// SupConLoss_76553497084189
// MI455X (gfx1250) — matrix operand order repaired and checked by reading, not compiled, not run on hardware
//
#include <hip/hip_runtime.h>
#include <math.h>

typedef __attribute__((ext_vector_type(16))) _Float16 v16h;
typedef __attribute__((ext_vector_type(8)))  _Float16 v8h;
typedef __attribute__((ext_vector_type(4)))  _Float16 v4h;
typedef __attribute__((ext_vector_type(8)))  float    v8f;

#define DIM 128
#define NCHUNK 16      // column chunks per 16-row tile in the full pass
#define NLAB 128       // label-space bound (labels in [0,100))

__device__ __forceinline__ int iminc(int a, int b) { return a < b ? a : b; }

// ---------------------------------------------------------------------------
// Row-normalize fp32 [n,128] -> fp16 [n,128]. One wave per row (wave32).
// ---------------------------------------------------------------------------
__global__ void __launch_bounds__(256) scl_normalize_f16(
    const float* __restrict__ src, _Float16* __restrict__ dst, int n) {
  const int wave = threadIdx.x >> 5;
  const int lane = threadIdx.x & 31;
  const int row  = blockIdx.x * 8 + wave;
  if (row >= n) return;
  float4 v = ((const float4*)(src + (size_t)row * DIM))[lane];
  float ss = v.x * v.x + v.y * v.y + v.z * v.z + v.w * v.w;
  #pragma unroll
  for (int off = 16; off; off >>= 1) ss += __shfl_xor(ss, off, 32);
  float rn = 1.0f / sqrtf(ss);
  v4h h;
  h[0] = (_Float16)(v.x * rn);
  h[1] = (_Float16)(v.y * rn);
  h[2] = (_Float16)(v.z * rn);
  h[3] = (_Float16)(v.w * rn);
  ((v4h*)(dst + (size_t)row * DIM))[lane] = h;
}

__global__ void scl_label_hist(const int* __restrict__ labels,
                               int* __restrict__ cnt, int n) {
  int i = blockIdx.x * blockDim.x + threadIdx.x;
  if (i < n) atomicAdd(&cnt[labels[i] & (NLAB - 1)], 1);
}

// Exclusive scan of 128 label counts (trivial serial; nanoseconds)
__global__ void scl_scan(const int* __restrict__ cnt, int* __restrict__ start,
                         int* __restrict__ cursor) {
  if (threadIdx.x == 0 && blockIdx.x == 0) {
    int s = 0;
    for (int l = 0; l < NLAB; ++l) { start[l] = s; cursor[l] = s; s += cnt[l]; }
  }
}

__global__ void scl_scatter(const int* __restrict__ labels,
                            int* __restrict__ cursor,
                            int* __restrict__ bucketRows, int n) {
  int i = blockIdx.x * blockDim.x + threadIdx.x;
  if (i < n) {
    int p = atomicAdd(&cursor[labels[i] & (NLAB - 1)], 1);
    bucketRows[p] = i;
  }
}

// ---------------------------------------------------------------------------
// Fragment helpers for V_WMMA_F32_16X16X32_F16 (this file's contiguous k order, same for A and B).
// A (16x32, MxK): lane l holds row l%16; halves = K in [16*(l/16),+16).
// B (32x16, KxN): lane l holds col l%16; halves = K in [16*(l/16),+16).
// C/D (16x16 f32): VGPR r -> row r + 8*(l/16), col l%16.
// ---------------------------------------------------------------------------
__device__ __forceinline__ v16h load_a_frag(const _Float16* arow, int kb, int hi) {
  union { v16h v; v8h h[2]; } u;
  u.h[0] = *(const v8h*)(arow + kb + hi * 16);
  u.h[1] = *(const v8h*)(arow + kb + hi * 16 + 8);
  return u.v;
}

// ---------------------------------------------------------------------------
// Pass 1: denom[i] += sum_j exp(s_ij)   (UNMASKED: no labels, no cndmask;
// positives are subtracted later by the tiny bucketed pass.)
// ---------------------------------------------------------------------------
__global__ void __launch_bounds__(256) scl_rowsum(
    const _Float16* __restrict__ Ah, const _Float16* __restrict__ Bh,
    float* __restrict__ denom, int n) {
  const int lane    = threadIdx.x & 31;
  const int gwave   = blockIdx.x * 8 + (threadIdx.x >> 5);
  const int rowTile = gwave / NCHUNK;
  const int chunk   = gwave % NCHUNK;
  const int i0 = rowTile * 16;
  const int colsPerChunk = n / NCHUNK;
  const int j0base = chunk * colsPerChunk;
  const int hi = lane >> 4;
  const int nn = lane & 15;

  const _Float16* arow = Ah + (size_t)(i0 + nn) * DIM;
  v16h a[4];
  #pragma unroll
  for (int k = 0; k < 4; ++k) a[k] = load_a_frag(arow, k * 32, hi);

  float acc[8] = {0.f, 0.f, 0.f, 0.f, 0.f, 0.f, 0.f, 0.f};

  for (int jt = 0; jt < colsPerChunk / 16; ++jt) {
    const int j0 = j0base + jt * 16;
    const _Float16* brow = Bh + (size_t)(j0 + nn) * DIM + hi * 16;
    v8f c = {};
    #pragma unroll
    for (int k = 0; k < 4; ++k) {
      v16h b = *(const v16h*)(brow + k * 32);
      c = __builtin_amdgcn_wmma_f32_16x16x32_f16(false, a[k], false, b,
                                                 (short)0, c, false, false);
    }
    #pragma unroll
    for (int r = 0; r < 8; ++r) acc[r] += __expf(c[r]);  // TEMPERATURE == 1
  }

  #pragma unroll
  for (int r = 0; r < 8; ++r) {
    float s = acc[r];
    s += __shfl_xor(s, 1, 32);
    s += __shfl_xor(s, 2, 32);
    s += __shfl_xor(s, 4, 32);
    s += __shfl_xor(s, 8, 32);          // reduce within each 16-lane half
    if (nn == 0) atomicAdd(&denom[i0 + 8 * hi + r], s);
  }
}

// ---------------------------------------------------------------------------
// Pass 2a (bucketed, ~0.7M pairs): denom[row] -= sum_{j in pos(row)} exp(s).
// Rows with the same label form the positive block; s is recomputed from the
// identical f16 operands so the subtraction cancels pass-1 terms exactly.
// One wave per (label, ti-stripe); ti strided for arbitrary bucket size.
// ---------------------------------------------------------------------------
__global__ void __launch_bounds__(256) scl_possub(
    const _Float16* __restrict__ Ah, const _Float16* __restrict__ Bh,
    const int* __restrict__ bucketStart, const int* __restrict__ labCnt,
    const int* __restrict__ bucketRows, float* __restrict__ denom) {
  const int lane  = threadIdx.x & 31;
  const int gwave = blockIdx.x * 8 + (threadIdx.x >> 5);
  const int L      = gwave >> 4;
  const int tiBase = gwave & 15;
  const int cnt   = labCnt[L];
  const int start = bucketStart[L];
  const int hi = lane >> 4;
  const int nn = lane & 15;

  for (int ti = tiBase; ti * 16 < cnt; ti += 16) {
    const int rposA = iminc(ti * 16 + nn, cnt - 1);
    const _Float16* arow = Ah + (size_t)bucketRows[start + rposA] * DIM;
    v16h a[4];
    #pragma unroll
    for (int k = 0; k < 4; ++k) a[k] = load_a_frag(arow, k * 32, hi);

    int rowpos[8], rowId[8];
    #pragma unroll
    for (int r = 0; r < 8; ++r) {
      rowpos[r] = ti * 16 + 8 * hi + r;
      rowId[r]  = bucketRows[start + iminc(rowpos[r], cnt - 1)];
    }

    float esum[8] = {0.f, 0.f, 0.f, 0.f, 0.f, 0.f, 0.f, 0.f};
    for (int tj = 0; tj * 16 < cnt; ++tj) {
      const int colpos = tj * 16 + nn;
      const _Float16* brow =
          Bh + (size_t)bucketRows[start + iminc(colpos, cnt - 1)] * DIM + hi * 16;
      v8f c = {};
      #pragma unroll
      for (int k = 0; k < 4; ++k) {
        v16h b = *(const v16h*)(brow + k * 32);
        c = __builtin_amdgcn_wmma_f32_16x16x32_f16(false, a[k], false, b,
                                                   (short)0, c, false, false);
      }
      const bool colOk = colpos < cnt;
      #pragma unroll
      for (int r = 0; r < 8; ++r) {
        float e = __expf(c[r]);
        if (colOk) esum[r] += e;
      }
    }

    #pragma unroll
    for (int r = 0; r < 8; ++r) {
      float s = esum[r];
      s += __shfl_xor(s, 1, 32);
      s += __shfl_xor(s, 2, 32);
      s += __shfl_xor(s, 4, 32);
      s += __shfl_xor(s, 8, 32);
      if (nn == 0 && rowpos[r] < cnt)
        denom[rowId[r]] -= s;           // single owner per row: plain RMW
    }
  }
}

// ---------------------------------------------------------------------------
// Pass 2b (bucketed): loss += sum_i (-1/cnt_i) * sum_{j in pos(i)}
//                     [ s_ij - log(exp(s_ij) + denom_i) ]
// ---------------------------------------------------------------------------
__global__ void __launch_bounds__(256) scl_posloss(
    const _Float16* __restrict__ Ah, const _Float16* __restrict__ Bh,
    const int* __restrict__ bucketStart, const int* __restrict__ labCnt,
    const int* __restrict__ bucketRows, const float* __restrict__ denom,
    float* __restrict__ out) {
  const int lane  = threadIdx.x & 31;
  const int gwave = blockIdx.x * 8 + (threadIdx.x >> 5);
  const int L      = gwave >> 4;
  const int tiBase = gwave & 15;
  const int cnt   = labCnt[L];
  const int start = bucketStart[L];
  const int hi = lane >> 4;
  const int nn = lane & 15;

  float part = 0.f;
  for (int ti = tiBase; ti * 16 < cnt; ti += 16) {
    const int rposA = iminc(ti * 16 + nn, cnt - 1);
    const _Float16* arow = Ah + (size_t)bucketRows[start + rposA] * DIM;
    v16h a[4];
    #pragma unroll
    for (int k = 0; k < 4; ++k) a[k] = load_a_frag(arow, k * 32, hi);

    int rowpos[8];
    float dn[8];
    #pragma unroll
    for (int r = 0; r < 8; ++r) {
      rowpos[r] = ti * 16 + 8 * hi + r;
      dn[r] = denom[bucketRows[start + iminc(rowpos[r], cnt - 1)]];
    }

    float psum[8] = {0.f, 0.f, 0.f, 0.f, 0.f, 0.f, 0.f, 0.f};
    for (int tj = 0; tj * 16 < cnt; ++tj) {
      const int colpos = tj * 16 + nn;
      const _Float16* brow =
          Bh + (size_t)bucketRows[start + iminc(colpos, cnt - 1)] * DIM + hi * 16;
      v8f c = {};
      #pragma unroll
      for (int k = 0; k < 4; ++k) {
        v16h b = *(const v16h*)(brow + k * 32);
        c = __builtin_amdgcn_wmma_f32_16x16x32_f16(false, a[k], false, b,
                                                   (short)0, c, false, false);
      }
      const bool colOk = colpos < cnt;
      #pragma unroll
      for (int r = 0; r < 8; ++r) {
        if (colOk) {
          float s = c[r];
          psum[r] += s - __logf(__expf(s) + dn[r]);
        }
      }
    }

    #pragma unroll
    for (int r = 0; r < 8; ++r) {
      float s = psum[r];
      s += __shfl_xor(s, 1, 32);
      s += __shfl_xor(s, 2, 32);
      s += __shfl_xor(s, 4, 32);
      s += __shfl_xor(s, 8, 32);
      if (nn == 0 && rowpos[r] < cnt) part += -s / (float)cnt;
    }
  }
  if (nn == 0 && part != 0.f) atomicAdd(out, part);
}

// ---------------------------------------------------------------------------
extern "C" void kernel_launch(void* const* d_in, const int* in_sizes, int n_in,
                              void* d_out, int out_size, void* d_ws, size_t ws_size,
                              hipStream_t stream) {
  const float* anchors   = (const float*)d_in[0];
  const float* positives = (const float*)d_in[1];
  const int*   labels    = (const int*)d_in[2];
  float*       out       = (float*)d_out;
  const int n = in_sizes[2];          // 8192

  _Float16* Ah     = (_Float16*)d_ws;
  _Float16* Bh     = Ah + (size_t)n * DIM;
  float*    denom  = (float*)(Bh + (size_t)n * DIM);
  int*      labCnt = (int*)(denom + n);
  int*      bStart = labCnt + NLAB;
  int*      cursor = bStart + NLAB;
  int*      bRows  = cursor + NLAB;

  hipMemsetAsync(denom, 0, (size_t)n * sizeof(float), stream);
  hipMemsetAsync(labCnt, 0, NLAB * sizeof(int), stream);
  hipMemsetAsync(out, 0, sizeof(float), stream);

  scl_normalize_f16<<<(n + 7) / 8, 256, 0, stream>>>(anchors, Ah, n);
  scl_normalize_f16<<<(n + 7) / 8, 256, 0, stream>>>(positives, Bh, n);
  scl_label_hist<<<(n + 255) / 256, 256, 0, stream>>>(labels, labCnt, n);
  scl_scan<<<1, 32, 0, stream>>>(labCnt, bStart, cursor);
  scl_scatter<<<(n + 255) / 256, 256, 0, stream>>>(labels, cursor, bRows, n);

  const int blocks1 = (n / 16) * NCHUNK / 8;     // 8 waves / 256-thread block
  scl_rowsum<<<blocks1, 256, 0, stream>>>(Ah, Bh, denom, n);

  const int blocks2 = NLAB * 16 / 8;             // (label, ti) waves
  scl_possub<<<blocks2, 256, 0, stream>>>(Ah, Bh, bStart, labCnt, bRows, denom);
  scl_posloss<<<blocks2, 256, 0, stream>>>(Ah, Bh, bStart, labCnt, bRows, denom, out);
}
